// SelfAttention_11673721110684
// MI455X (gfx1250) — matrix operand order repaired and checked by reading, not compiled, not run on hardware
//
#include <hip/hip_runtime.h>

// MI455X / gfx1250, wave32. Self-attention B=4,C=128,H=W=64 (N=4096), 4 heads x 32.
// bf16 WMMA (v_wmma_f32_16x16x32_bf16), f32 accumulate, flash-attention (scores
// never materialized). All LDS tiles are staged "fragment-ready" so every WMMA
// operand is fetched with two ds_load_b128 per lane (no scalar u16 gathers).

typedef __attribute__((ext_vector_type(16))) __bf16 v16bf;
typedef __attribute__((ext_vector_type(8)))  __bf16 v8bf;
typedef __attribute__((ext_vector_type(8)))  float  v8f;

__device__ __forceinline__ v8f wmma_bf16(v16bf a, v16bf b, v8f c) {
  return __builtin_amdgcn_wmma_f32_16x16x32_bf16(false, a, false, b, (short)0, c,
                                                 false, false);
}

// A-matrix 16x32 bf16 (ISA 7.12.2): lane m = lane&15, kb = (lane&16)?8:0,
// elements = row m, K runs [kb..kb+7] and [kb+16..kb+23] -> two 16B loads.
// base must be 16B aligned, ld (elements) a multiple of 8.
__device__ __forceinline__ v16bf frag_a16x32(const __bf16* base, int ld) {
  const int lane = threadIdx.x & 31;
  const __bf16* p = base + (lane & 15) * ld + ((lane & 16) ? 8 : 0);
  const v8bf lo = *(const v8bf*)(p);
  const v8bf hi = *(const v8bf*)(p + 16);
  return __builtin_shufflevector(lo, hi, 0, 1, 2, 3, 4, 5, 6, 7, 8, 9, 10, 11,
                                 12, 13, 14, 15);
}

// B-matrix 32x16 bf16: lane col n = lane&15, K = [kb..kb+7],[kb+16..kb+23] (kb = 0|8).
// Tiles are staged column-major (base[n*ld + K]) -> two contiguous 8-element runs per lane.
__device__ __forceinline__ v16bf frag_b32x16(const __bf16* base, int ld) {
  const int lane = threadIdx.x & 31;
  const __bf16* p = base + (lane & 15) * ld + ((lane & 16) ? 8 : 0);
  const v8bf lo = *(const v8bf*)(p);
  const v8bf hi = *(const v8bf*)(p + 16);
  return __builtin_shufflevector(lo, hi, 0, 1, 2, 3, 4, 5, 6, 7, 8, 9, 10, 11,
                                 12, 13, 14, 15);
}

__device__ __forceinline__ float half_max16(float v) {
  v = fmaxf(v, __shfl_xor(v, 1, 32));
  v = fmaxf(v, __shfl_xor(v, 2, 32));
  v = fmaxf(v, __shfl_xor(v, 4, 32));
  v = fmaxf(v, __shfl_xor(v, 8, 32));
  return v;
}
__device__ __forceinline__ float half_sum16(float v) {
  v += __shfl_xor(v, 1, 32);
  v += __shfl_xor(v, 2, 32);
  v += __shfl_xor(v, 4, 32);
  v += __shfl_xor(v, 8, 32);
  return v;
}

// ---------------- stage 0: fp32 -> bf16 convert (x4 vectorized) ----------------
__global__ void cvt_f32_bf16_kernel(const float* __restrict__ in,
                                    __bf16* __restrict__ out, int n4) {
  const int i = blockIdx.x * 256 + threadIdx.x;
  if (i < n4) {
    const float4 f = ((const float4*)in)[i];
    union { uint2 u; __bf16 h[4]; } o;
    o.h[0] = (__bf16)f.x; o.h[1] = (__bf16)f.y;
    o.h[2] = (__bf16)f.z; o.h[3] = (__bf16)f.w;
    ((uint2*)out)[i] = o.u;
  }
}

// ---------------- stage 1: fused q/k/v 1x1-conv projection ----------------
// grid = B*(N/128), block = 256 (8 waves). Weight tile row-major [o][c] for
// A-frags; x tile token-major [n][c] (padded ld=136) for B-frags.
__global__ __launch_bounds__(256) void qkv_proj_kernel(
    const __bf16* __restrict__ xb, const __bf16* __restrict__ wb,
    const float* __restrict__ bq, const float* __restrict__ bk,
    const float* __restrict__ bv, __bf16* __restrict__ qo,
    __bf16* __restrict__ ko, __bf16* __restrict__ vo) {
  constexpr int C = 128, N = 4096, XLD = 136;
  const int tile = blockIdx.x & 31;
  const int b    = blockIdx.x >> 5;
  const int n0   = tile * 128;
  const int tid  = threadIdx.x;
  const int wave = tid >> 5, lane = tid & 31;
  const int hi8  = (lane & 16) ? 8 : 0;
  const int nc   = lane & 15;
  const int o0r  = wave * 16;

  __shared__ __align__(16) __bf16 xs[128 * XLD];  // [token][c], ~34 KB
  __shared__ __align__(16) __bf16 wsm[C * C];     // [o][c], 32 KB

  for (int idx = tid; idx < C * 128; idx += 256) {
    const int c = idx >> 7, nn = idx & 127;       // coalesced along nn
    xs[nn * XLD + c] = xb[((size_t)b * C + c) * N + n0 + nn];
  }

#pragma unroll 1
  for (int mat = 0; mat < 3; ++mat) {
    const __bf16* wsrc = wb + mat * C * C;
    const float*  bias = (mat == 0) ? bq : ((mat == 1) ? bk : bv);
    __bf16*       outp = (mat == 0) ? qo : ((mat == 1) ? ko : vo);
    __syncthreads();  // previous mat's wmma reads of wsm done
    for (int idx = tid; idx < (C * C) / 8; idx += 256)
      ((v8bf*)wsm)[idx] = ((const v8bf*)wsrc)[idx];  // b128 in / b128 out
    __syncthreads();

#pragma unroll 1
    for (int j = 0; j < 8; ++j) {  // 8 column sub-tiles of 16 tokens
      v8f acc = {};
#pragma unroll
      for (int kc = 0; kc < 4; ++kc) {  // C=128 in K=32 steps
        const v16bf a  = frag_a16x32(wsm + o0r * C + kc * 32, C);
        const v16bf bm = frag_b32x16(xs + (j * 16) * XLD + kc * 32, XLD);
        acc = wmma_bf16(a, bm, acc);
      }
#pragma unroll
      for (int r = 0; r < 8; ++r) {
        const int oc = o0r + r + hi8;
        const float val = acc[r] + bias[oc];
        outp[((size_t)b * C + oc) * N + n0 + j * 16 + nc] = (__bf16)val;
      }
    }
  }
}

// ---------------- stage 2: flash attention ----------------
// grid = B*NH*(N/128), block = 256 (8 waves x 16 queries). Online softmax,
// 32-key chunks staged in LDS (K key-major, V d-major: fragment-ready).
__global__ __launch_bounds__(256) void flash_attn_kernel(
    const __bf16* __restrict__ q, const __bf16* __restrict__ k,
    const __bf16* __restrict__ v, __bf16* __restrict__ o) {
  constexpr int C = 128, N = 4096, HD = 32, TLD = 40;  // padded tile ld
  const int tile = blockIdx.x & 31;
  const int bh   = blockIdx.x >> 5;
  const int b = bh >> 2, h = bh & 3;
  const int n0 = tile * 128;
  const int tid = threadIdx.x;
  const int wave = tid >> 5, lane = tid & 31;
  const int m0  = wave * 16;
  const int hi8 = (lane & 16) ? 8 : 0;
  const int nc  = lane & 15;

  const __bf16* qg = q + ((size_t)b * C + h * HD) * N;
  const __bf16* kg = k + ((size_t)b * C + h * HD) * N;
  const __bf16* vg = v + ((size_t)b * C + h * HD) * N;

  __shared__ __align__(16) __bf16 lq[128 * TLD];     // [query][d]  (A-ready)
  __shared__ __align__(16) __bf16 lk[32 * TLD];      // [key][d]    (B-ready, K=d)
  __shared__ __align__(16) __bf16 lv[32 * TLD];      // [d][key]    (B-ready, K=key)
  __shared__ __align__(16) __bf16 lp[8 * 16 * TLD];  // per-wave P  (A-ready)
  __bf16* lpw = lp + wave * 16 * TLD;

  for (int idx = tid; idx < 128 * 32; idx += 256) {
    const int nn = idx & 127, d = idx >> 7;          // coalesced global read
    lq[nn * TLD + d] = qg[(size_t)d * N + n0 + nn];
  }
  __syncthreads();

  const v16bf aq = frag_a16x32(lq + m0 * TLD, TLD);  // 16 queries x 32 d

  v8f o0 = {}, o1 = {};
  float mrow[8], lrow[8];
#pragma unroll
  for (int r = 0; r < 8; ++r) { mrow[r] = -3.0e38f; lrow[r] = 0.f; }

  const float scale = 0.17677669529663687f;  // 1/sqrt(32)
  const int ld_d = tid >> 3;                 // chunk loader: d row 0..31
  const int ld_c = (tid & 7) * 4;            // 4 contiguous keys

#pragma unroll 1
  for (int kc = 0; kc < N / 32; ++kc) {
    const int nk = kc * 32;
    __syncthreads();  // all waves done with previous chunk
    {
      const __bf16* ks = kg + (size_t)ld_d * N + nk + ld_c;
      const __bf16* vs = vg + (size_t)ld_d * N + nk + ld_c;
      union { uint2 u; __bf16 h[4]; } uk;
      uk.u = *(const uint2*)ks;                       // b64 global load
#pragma unroll
      for (int i = 0; i < 4; ++i)
        lk[(ld_c + i) * TLD + ld_d] = uk.h[i];        // transpose -> [key][d]
      *(uint2*)&lv[ld_d * TLD + ld_c] = *(const uint2*)vs;  // direct b64 copy
      if (kc + 1 < N / 32) {                          // global_prefetch_b8
        __builtin_prefetch(ks + 32, 0, 1);
        __builtin_prefetch(vs + 32, 0, 1);
      }
    }
    __syncthreads();

    v8f zero = {};
    v8f s0 = wmma_bf16(aq, frag_b32x16(lk, TLD), zero);            // keys 0..15
    v8f s1 = wmma_bf16(aq, frag_b32x16(lk + 16 * TLD, TLD), zero); // keys 16..31

    // online softmax; D-layout row = r + hi8, stats replicated per 16-lane half
#pragma unroll
    for (int r = 0; r < 8; ++r) {
      const float x0 = s0[r] * scale;
      const float x1 = s1[r] * scale;
      const float mx   = half_max16(fmaxf(x0, x1));
      const float mnew = fmaxf(mrow[r], mx);
      const float corr = __expf(mrow[r] - mnew);
      const float p0 = __expf(x0 - mnew);
      const float p1 = __expf(x1 - mnew);
      lrow[r] = lrow[r] * corr + half_sum16(p0 + p1);
      mrow[r] = mnew;
      o0[r] *= corr;
      o1[r] *= corr;
      lpw[(r + hi8) * TLD + nc]      = (__bf16)p0;  // D-layout -> A-layout
      lpw[(r + hi8) * TLD + 16 + nc] = (__bf16)p1;
    }

    const v16bf ap = frag_a16x32(lpw, TLD);                    // 16 q x 32 keys
    o0 = wmma_bf16(ap, frag_b32x16(lv, TLD), o0);              // d 0..15
    o1 = wmma_bf16(ap, frag_b32x16(lv + 16 * TLD, TLD), o1);   // d 16..31
  }

  __bf16* og = o + ((size_t)b * C + h * HD) * N;
#pragma unroll
  for (int r = 0; r < 8; ++r) {
    const float inv = 1.0f / lrow[r];
    const int nq = n0 + m0 + r + hi8;
    og[(size_t)nc * N + nq]        = (__bf16)(o0[r] * inv);
    og[(size_t)(nc + 16) * N + nq] = (__bf16)(o1[r] * inv);
  }
}

// ---------------- stage 3: output projection + bias + residual ----------------
__global__ __launch_bounds__(256) void out_proj_kernel(
    const __bf16* __restrict__ ab, const __bf16* __restrict__ wo,
    const float* __restrict__ bo, const float* __restrict__ xres,
    float* __restrict__ out) {
  constexpr int C = 128, N = 4096, XLD = 136;
  const int tile = blockIdx.x & 31;
  const int b    = blockIdx.x >> 5;
  const int n0   = tile * 128;
  const int tid  = threadIdx.x;
  const int wave = tid >> 5, lane = tid & 31;
  const int hi8  = (lane & 16) ? 8 : 0;
  const int nc   = lane & 15;
  const int o0r  = wave * 16;

  __shared__ __align__(16) __bf16 as_[128 * XLD];  // [token][c]
  __shared__ __align__(16) __bf16 wsm[C * C];      // [o][c]

  for (int idx = tid; idx < C * 128; idx += 256) {
    const int c = idx >> 7, nn = idx & 127;
    as_[nn * XLD + c] = ab[((size_t)b * C + c) * N + n0 + nn];
  }
  for (int idx = tid; idx < (C * C) / 8; idx += 256)
    ((v8bf*)wsm)[idx] = ((const v8bf*)wo)[idx];
  __syncthreads();

#pragma unroll 1
  for (int j = 0; j < 8; ++j) {
    v8f acc = {};
#pragma unroll
    for (int kc = 0; kc < 4; ++kc) {
      const v16bf a  = frag_a16x32(wsm + o0r * C + kc * 32, C);
      const v16bf bm = frag_b32x16(as_ + (j * 16) * XLD + kc * 32, XLD);
      acc = wmma_bf16(a, bm, acc);
    }
#pragma unroll
    for (int r = 0; r < 8; ++r) {
      const int oc = o0r + r + hi8;
      const size_t gi = ((size_t)b * C + oc) * N + n0 + j * 16 + nc;
      out[gi] = acc[r] + bo[oc] + xres[gi];
    }
  }
}

// ---------------- host launcher ----------------
extern "C" void kernel_launch(void* const* d_in, const int* in_sizes, int n_in,
                              void* d_out, int out_size, void* d_ws, size_t ws_size,
                              hipStream_t stream) {
  (void)in_sizes; (void)n_in; (void)out_size; (void)ws_size;
  constexpr int B = 4, C = 128, N = 4096;
  const size_t nx = (size_t)B * C * N;  // 2,097,152

  const float* x  = (const float*)d_in[0];
  const float* wq = (const float*)d_in[1];
  const float* bq = (const float*)d_in[2];
  const float* wk = (const float*)d_in[3];
  const float* bk = (const float*)d_in[4];
  const float* wv = (const float*)d_in[5];
  const float* bv = (const float*)d_in[6];
  const float* wo = (const float*)d_in[7];
  const float* bo = (const float*)d_in[8];

  // workspace carve-up (bf16): x, 4 weights, q, k, v, attn  ~= 21.2 MB
  __bf16* xb = (__bf16*)d_ws;
  __bf16* wb = xb + nx;
  __bf16* qb = wb + 4 * C * C;
  __bf16* kb = qb + nx;
  __bf16* vb = kb + nx;
  __bf16* abuf = vb + nx;

  cvt_f32_bf16_kernel<<<(int)(nx / 4 / 256), 256, 0, stream>>>(x, xb, (int)(nx / 4));
  cvt_f32_bf16_kernel<<<16, 256, 0, stream>>>(wq, wb + 0 * C * C, C * C / 4);
  cvt_f32_bf16_kernel<<<16, 256, 0, stream>>>(wk, wb + 1 * C * C, C * C / 4);
  cvt_f32_bf16_kernel<<<16, 256, 0, stream>>>(wv, wb + 2 * C * C, C * C / 4);
  cvt_f32_bf16_kernel<<<16, 256, 0, stream>>>(wo, wb + 3 * C * C, C * C / 4);

  qkv_proj_kernel<<<B * 32, 256, 0, stream>>>(xb, wb, bq, bk, bv, qb, kb, vb);
  flash_attn_kernel<<<B * 4 * 32, 256, 0, stream>>>(qb, kb, vb, abuf);
  out_proj_kernel<<<B * 32, 256, 0, stream>>>(abuf, wb + 3 * C * C, bo, x,
                                              (float*)d_out);
}
